// EncoderEGNCA_52381421142728
// MI455X (gfx1250) — matrix operand order repaired and checked by reading, not compiled, not run on hardware
//
#include <hip/hip_runtime.h>
#include <hip/hip_bf16.h>
#include <math.h>

typedef _Float16 half_t;
typedef __attribute__((ext_vector_type(16))) _Float16 v16h;
typedef __attribute__((ext_vector_type(8)))  _Float16 v8h;
typedef __attribute__((ext_vector_type(8)))  float    v8f;

#define WAVES 4          // 128-thread blocks = 4 wave32
#define EPS 1e-5f

// ---------------------------------------------------------------------------
// WMMA + math helpers
// ---------------------------------------------------------------------------
__device__ __forceinline__ v8f wmma16(v16h a, v16h b, v8f c) {
  return __builtin_amdgcn_wmma_f32_16x16x32_f16(false, a, false, b, (short)0, c,
                                                false, false);
}

__device__ __forceinline__ void lds_fence() {
  asm volatile("s_wait_dscnt 0" ::: "memory");
}

// Branch-free tanh: tanh(|x|) = (1 - e) / (1 + e), e = exp(-2|x|) via v_exp_f32.
__device__ __forceinline__ float fast_tanh(float x) {
  float ax = __builtin_fabsf(x);
  float e  = __builtin_amdgcn_exp2f(ax * -2.88539008178f);   // exp(-2|x|)
  float t  = (1.f - e) * __builtin_amdgcn_rcpf(1.f + e);
  return __builtin_copysignf(t, x);
}

__device__ __forceinline__ v8f splat8(float v) {
  v8f c;
#pragma unroll
  for (int i = 0; i < 8; ++i) c[i] = v;
  return c;
}

// B fragment (K=32 x N=16 slice of row-major W[Krows x Ncols], zero-padded).
// lane l holds column n = nbase + (l&15); lane-half selects K 0-15 / 16-31.
__device__ __forceinline__ v16h load_b_frag(const float* W, int Krows, int Ncols,
                                            int kbase, int nbase, int lane) {
  v16h f;
  int n  = nbase + (lane & 15);
  int kh = lane >> 4;
#pragma unroll
  for (int j = 0; j < 16; ++j) {
    int k = kbase + kh * 16 + j;
    float v = (k < Krows && n < Ncols) ? W[k * Ncols + n] : 0.f;
    f[j] = (half_t)v;
  }
  return f;
}

// A fragment from LDS f16 tile [16 rows][stride halfs], K slice kbase..kbase+31.
// lane l: row m = l&15; elems 0-15 = K 16*kh.. (contiguous, matches load_b_frag)
__device__ __forceinline__ v16h load_a_frag(const half_t* tile, int stride,
                                            int kbase, int lane) {
  int m = lane & 15, kh = lane >> 4;
  v16h f;
#pragma unroll
  for (int j = 0; j < 16; ++j) {
    int k = kbase + 16 * kh + j;
    f[j] = tile[m * stride + k];
  }
  return f;
}

// Store D (C-layout: lane holds column n, rows mb..mb+7) with tanh, f16.
__device__ __forceinline__ void store_d_tanh(half_t* tile, int stride, int nbase,
                                             int lane, v8f d) {
  int n = nbase + (lane & 15);
  int mb = 8 * (lane >> 4);
#pragma unroll
  for (int v = 0; v < 8; ++v) tile[(mb + v) * stride + n] = (half_t)fast_tanh(d[v]);
}

__device__ __forceinline__ void fatomic_add(float* p, float v) {
  __hip_atomic_fetch_add(p, v, __ATOMIC_RELAXED, __HIP_MEMORY_SCOPE_AGENT);
}

__device__ __forceinline__ v8h pack8(float4 a, float4 b) {
  v8h r;
  r[0] = (half_t)a.x; r[1] = (half_t)a.y; r[2] = (half_t)a.z; r[3] = (half_t)a.w;
  r[4] = (half_t)b.x; r[5] = (half_t)b.y; r[6] = (half_t)b.z; r[7] = (half_t)b.w;
  return r;
}

// ---------------------------------------------------------------------------
// Workspace zeroing
// ---------------------------------------------------------------------------
__global__ void egnca_zero(float4* p, long n4) {
  long i = (long)blockIdx.x * blockDim.x + threadIdx.x;
  long s = (long)gridDim.x * blockDim.x;
  float4 z = {0.f, 0.f, 0.f, 0.f};
  for (; i < n4; i += s) p[i] = z;
}

// ---------------------------------------------------------------------------
// Edge kernel: message MLP + coord weight, scatter-add into num/cnt/agg.
// One wave per 16-edge tile.
// ---------------------------------------------------------------------------
__global__ void __launch_bounds__(WAVES * 32, 1)
egnca_edge(const int* __restrict__ ei, const float* __restrict__ coord,
           const float* __restrict__ nf,
           const float* __restrict__ mw1, const float* __restrict__ mb1,
           const float* __restrict__ mw2, const float* __restrict__ mb2,
           const float* __restrict__ cw1, const float* __restrict__ cb1,
           const float* __restrict__ cw2,
           float* __restrict__ num, float* __restrict__ cnt,
           float* __restrict__ agg, int Nn, int En) {
  __shared__ half_t s_ein[WAVES][16 * 64];   // edge_in tile, K padded to 64
  __shared__ half_t s_act[WAVES][16 * 32];   // activation bounce tile
  __shared__ float  s_m32[WAVES][16 * 32];   // message (f32) for scatter
  __shared__ int    s_row[WAVES][16];
  __shared__ float  s_cw2[32];

  const int lane = threadIdx.x & 31;
  const int wave = threadIdx.x >> 5;
  half_t* ein  = s_ein[wave];
  half_t* actb = s_act[wave];
  float*  m32  = s_m32[wave];
  int*    rowa = s_row[wave];

  if (threadIdx.x < 32) s_cw2[threadIdx.x] = cw2[threadIdx.x];
  __syncthreads();

  // ---- per-wave weight fragments + per-lane bias scalars (loop invariant) ----
  v16h B1[2][2], B2[2], Bc1[2];
#pragma unroll
  for (int kc = 0; kc < 2; ++kc)
#pragma unroll
    for (int nc = 0; nc < 2; ++nc)
      B1[kc][nc] = load_b_frag(mw1, 33, 32, kc * 32, nc * 16, lane);
#pragma unroll
  for (int nc = 0; nc < 2; ++nc) {
    B2[nc]  = load_b_frag(mw2, 32, 32, 0, nc * 16, lane);
    Bc1[nc] = load_b_frag(cw1, 32, 32, 0, nc * 16, lane);
  }
  const int nn = lane & 15;
  const float bm1a = mb1[nn], bm1b = mb1[16 + nn];
  const float bm2a = mb2[nn], bm2b = mb2[16 + nn];
  const float bc1a = cb1[nn], bc1b = cb1[16 + nn];

  const int tiles  = (En + 15) >> 4;
  const int nwaves = gridDim.x * (blockDim.x >> 5);
  const int wid    = blockIdx.x * (blockDim.x >> 5) + wave;
  const v8h zero8  = {};

  for (int t = wid; t < tiles; t += nwaves) {
    const int ebase = t << 4;
    float rx = 0.f, ry = 0.f, rz = 0.f;
    int myrow = 0;
    bool valid = false;

    if (lane < 16) {
      int eg = ebase + lane;
      valid = eg < En;
      int e = valid ? eg : (En - 1);
      int r = ei[e];
      int c = ei[En + e];
      myrow = r;
      rowa[lane] = valid ? r : -1;
      rx = coord[r * 3 + 0] - coord[c * 3 + 0];
      ry = coord[r * 3 + 1] - coord[c * 3 + 1];
      rz = coord[r * 3 + 2] - coord[c * 3 + 2];
      float dist = rx * rx + ry * ry + rz * rz;
      const float4* nr = (const float4*)(nf + (size_t)r * 16);
      const float4* nc = (const float4*)(nf + (size_t)c * 16);
      float4 r0 = nr[0], r1 = nr[1], r2 = nr[2], r3 = nr[3];
      float4 c0 = nc[0], c1 = nc[1], c2 = nc[2], c3 = nc[3];
      v8h* dst = (v8h*)(ein + lane * 64);
      dst[0] = pack8(r0, r1);
      dst[1] = pack8(r2, r3);
      dst[2] = pack8(c0, c1);
      dst[3] = pack8(c2, c3);
      v8h dv = zero8; dv[0] = (half_t)dist;
      dst[4] = dv;
      dst[5] = zero8; dst[6] = zero8; dst[7] = zero8;
    }
    lds_fence();

    // ---- message layer 1: [16x64(33)] x [64x32] ----
    v16h a0 = load_a_frag(ein, 64, 0, lane);
    v16h a1 = load_a_frag(ein, 64, 32, lane);
    v8f d0 = splat8(bm1a);
    v8f d1 = splat8(bm1b);
    d0 = wmma16(a0, B1[0][0], d0); d0 = wmma16(a1, B1[1][0], d0);
    d1 = wmma16(a0, B1[0][1], d1); d1 = wmma16(a1, B1[1][1], d1);
    store_d_tanh(actb, 32, 0, lane, d0);
    store_d_tanh(actb, 32, 16, lane, d1);
    lds_fence();

    // ---- message layer 2: [16x32] x [32x32] ----
    v16h am = load_a_frag(actb, 32, 0, lane);
    v8f e0 = splat8(bm2a);
    v8f e1 = splat8(bm2b);
    e0 = wmma16(am, B2[0], e0);
    e1 = wmma16(am, B2[1], e1);
    // m = tanh(e): f32 copy for scatter, f16 copy for coord MLP input
    {
      int mb = 8 * (lane >> 4);
#pragma unroll
      for (int v = 0; v < 8; ++v) {
        float m0 = fast_tanh(e0[v]);
        float m1 = fast_tanh(e1[v]);
        actb[(mb + v) * 32 + nn]      = (half_t)m0;
        actb[(mb + v) * 32 + 16 + nn] = (half_t)m1;
        m32[(mb + v) * 32 + nn]       = m0;
        m32[(mb + v) * 32 + 16 + nn]  = m1;
      }
    }
    lds_fence();

    // ---- coord MLP layer 1: tanh(m @ cw1 + cb1) ----
    v16h ac = load_a_frag(actb, 32, 0, lane);
    v8f h0 = splat8(bc1a);
    v8f h1 = splat8(bc1b);
    h0 = wmma16(ac, Bc1[0], h0);
    h1 = wmma16(ac, Bc1[1], h1);
    store_d_tanh(actb, 32, 0, lane, h0);
    store_d_tanh(actb, 32, 16, lane, h1);
    lds_fence();

    // ---- coord weight (32 -> 1) + num/cnt scatter (lanes 0..15 = edges) ----
    if (lane < 16 && valid) {
      float acc = 0.f;
#pragma unroll
      for (int f = 0; f < 32; ++f) acc += (float)actb[lane * 32 + f] * s_cw2[f];
      float cwv = fast_tanh(acc);
      fatomic_add(&num[myrow * 3 + 0], rx * cwv);
      fatomic_add(&num[myrow * 3 + 1], ry * cwv);
      fatomic_add(&num[myrow * 3 + 2], rz * cwv);
      fatomic_add(&cnt[myrow], 1.f);
    }

    // ---- agg scatter: 32 lanes (features) x 16 edges ----
#pragma unroll 4
    for (int e2 = 0; e2 < 16; ++e2) {
      int r = rowa[e2];
      if (r >= 0) fatomic_add(&agg[r * 32 + lane], m32[e2 * 32 + lane]);
    }
  }
}

// ---------------------------------------------------------------------------
// Node kernel: coord mean-update + node MLP + residual + NodeNorm.
// One wave per 16-node tile.
// ---------------------------------------------------------------------------
__global__ void __launch_bounds__(WAVES * 32, 1)
egnca_node(const float* __restrict__ coord, const float* __restrict__ nf,
           const float* __restrict__ nw1, const float* __restrict__ nb1,
           const float* __restrict__ nw2, const float* __restrict__ nb2,
           const float* __restrict__ num, const float* __restrict__ cnt,
           const float* __restrict__ agg,
           float* __restrict__ out_coord, float* __restrict__ out_h, int Nn) {
  __shared__ half_t s_cat[WAVES][16 * 64];   // concat(nf, agg) padded to 64
  __shared__ half_t s_act[WAVES][16 * 32];
  __shared__ float  s_upd[WAVES][16 * 16];

  const int lane = threadIdx.x & 31;
  const int wave = threadIdx.x >> 5;
  half_t* cat  = s_cat[wave];
  half_t* actb = s_act[wave];
  float*  upd  = s_upd[wave];

  v16h Bn1[2][2], Bn2;
#pragma unroll
  for (int kc = 0; kc < 2; ++kc)
#pragma unroll
    for (int nc = 0; nc < 2; ++nc)
      Bn1[kc][nc] = load_b_frag(nw1, 48, 32, kc * 32, nc * 16, lane);
  Bn2 = load_b_frag(nw2, 32, 16, 0, 0, lane);

  const int nn = lane & 15;
  const float bn1a = nb1[nn], bn1b = nb1[16 + nn];
  const float bn2a = nb2[nn];

  const int tiles  = (Nn + 15) >> 4;
  const int nwaves = gridDim.x * (blockDim.x >> 5);
  const int wid    = blockIdx.x * (blockDim.x >> 5) + wave;
  const v8h zero8  = {};

  for (int t = wid; t < tiles; t += nwaves) {
    const int nbase = t << 4;
    float nfv[16];
    bool valid = false;

    if (lane < 16) {
      int ng = nbase + lane;
      valid = ng < Nn;
      int nid = valid ? ng : (Nn - 1);
      // coord mean update (straight to output)
      float c = cnt[nid];
      float inv = __builtin_amdgcn_rcpf(fmaxf(c, 1.f));
      if (valid) {
#pragma unroll
        for (int d = 0; d < 3; ++d)
          out_coord[nid * 3 + d] = coord[nid * 3 + d] + num[nid * 3 + d] * inv;
      }
      const float4* nr = (const float4*)(nf  + (size_t)nid * 16);
      const float4* ag = (const float4*)(agg + (size_t)nid * 32);
      float4 q0 = nr[0], q1 = nr[1], q2 = nr[2], q3 = nr[3];
      float4 g0 = ag[0], g1 = ag[1], g2 = ag[2], g3 = ag[3];
      float4 g4 = ag[4], g5 = ag[5], g6 = ag[6], g7 = ag[7];
      nfv[0]=q0.x; nfv[1]=q0.y; nfv[2]=q0.z; nfv[3]=q0.w;
      nfv[4]=q1.x; nfv[5]=q1.y; nfv[6]=q1.z; nfv[7]=q1.w;
      nfv[8]=q2.x; nfv[9]=q2.y; nfv[10]=q2.z; nfv[11]=q2.w;
      nfv[12]=q3.x; nfv[13]=q3.y; nfv[14]=q3.z; nfv[15]=q3.w;
      v8h* dst = (v8h*)(cat + lane * 64);
      dst[0] = pack8(q0, q1);
      dst[1] = pack8(q2, q3);
      dst[2] = pack8(g0, g1);
      dst[3] = pack8(g2, g3);
      dst[4] = pack8(g4, g5);
      dst[5] = pack8(g6, g7);
      dst[6] = zero8; dst[7] = zero8;
    }
    lds_fence();

    // ---- node layer 1: [16x64(48)] x [64x32] ----
    v16h a0 = load_a_frag(cat, 64, 0, lane);
    v16h a1 = load_a_frag(cat, 64, 32, lane);
    v8f d0 = splat8(bn1a);
    v8f d1 = splat8(bn1b);
    d0 = wmma16(a0, Bn1[0][0], d0); d0 = wmma16(a1, Bn1[1][0], d0);
    d1 = wmma16(a0, Bn1[0][1], d1); d1 = wmma16(a1, Bn1[1][1], d1);
    store_d_tanh(actb, 32, 0, lane, d0);
    store_d_tanh(actb, 32, 16, lane, d1);
    lds_fence();

    // ---- node layer 2: [16x32] x [32x16] ----
    v16h a2 = load_a_frag(actb, 32, 0, lane);
    v8f u = splat8(bn2a);
    u = wmma16(a2, Bn2, u);
    {
      int mb = 8 * (lane >> 4);
#pragma unroll
      for (int v = 0; v < 8; ++v) upd[(mb + v) * 16 + nn] = u[v];
    }
    lds_fence();

    // ---- residual + NodeNorm (root_power 2 => h / sqrt(std)) ----
    if (lane < 16 && valid) {
      int nid = nbase + lane;
      float h[16];
      float mean = 0.f;
#pragma unroll
      for (int f = 0; f < 16; ++f) {
        h[f] = nfv[f] + upd[lane * 16 + f];
        mean += h[f];
      }
      mean *= (1.f / 16.f);
      float var = 0.f;
#pragma unroll
      for (int f = 0; f < 16; ++f) {
        float d = h[f] - mean;
        var += d * d;
      }
      var *= (1.f / 16.f);
      float scale = rsqrtf(sqrtf(var + EPS));
      float4* oh = (float4*)(out_h + (size_t)nid * 16);
#pragma unroll
      for (int q = 0; q < 4; ++q) {
        float4 o;
        o.x = h[q * 4 + 0] * scale; o.y = h[q * 4 + 1] * scale;
        o.z = h[q * 4 + 2] * scale; o.w = h[q * 4 + 3] * scale;
        oh[q] = o;
      }
    }
  }
}

// ---------------------------------------------------------------------------
// Launcher
// ---------------------------------------------------------------------------
extern "C" void kernel_launch(void* const* d_in, const int* in_sizes, int n_in,
                              void* d_out, int out_size, void* d_ws, size_t ws_size,
                              hipStream_t stream) {
  const int*   ei    = (const int*)d_in[0];
  const float* coord = (const float*)d_in[1];
  const float* nfeat = (const float*)d_in[2];
  const float* mw1   = (const float*)d_in[3];
  const float* mb1   = (const float*)d_in[4];
  const float* mw2   = (const float*)d_in[5];
  const float* mb2   = (const float*)d_in[6];
  const float* cw1   = (const float*)d_in[7];
  const float* cb1   = (const float*)d_in[8];
  const float* cw2   = (const float*)d_in[9];
  const float* nw1   = (const float*)d_in[10];
  const float* nb1   = (const float*)d_in[11];
  const float* nw2   = (const float*)d_in[12];
  const float* nb2   = (const float*)d_in[13];
  // d_in[14] = n_steps (1 in this config)

  const int E = in_sizes[0] / 2;
  const int N = in_sizes[1] / 3;

  float* ws  = (float*)d_ws;
  float* num = ws;             // N*3
  float* cnt = ws + 3 * N;     // N
  float* agg = ws + 4 * N;     // N*32 ; total 36*N floats (divisible by 4)

  float* out_coord = (float*)d_out;
  float* out_h     = out_coord + (long)N * 3;

  egnca_zero<<<512, 256, 0, stream>>>((float4*)ws, (long)N * 36 / 4);

  const int threads = WAVES * 32;
  int egrid = (E / 16 + WAVES - 1) / WAVES;
  if (egrid > 4096) egrid = 4096;
  egnca_edge<<<egrid, threads, 0, stream>>>(ei, coord, nfeat, mw1, mb1, mw2, mb2,
                                            cw1, cb1, cw2, num, cnt, agg, N, E);

  int ngrid = (N / 16 + WAVES - 1) / WAVES;
  if (ngrid > 2048) ngrid = 2048;
  egnca_node<<<ngrid, threads, 0, stream>>>(coord, nfeat, nw1, nb1, nw2, nb2,
                                            num, cnt, agg, out_coord, out_h, N);
}
